// NeuralCDE_48490180771849
// MI455X (gfx1250) — matrix operand order repaired and checked by reading, not compiled, not run on hardware
//
#include <hip/hip_runtime.h>
#include <hip/hip_bf16.h>

#define B_  512
#define T_  256
#define IN_ 32
#define H_  128
#define HH_ 128
#define OUT_ 64

typedef unsigned int u32;
typedef __attribute__((ext_vector_type(16))) __bf16 v16bf;
typedef __attribute__((ext_vector_type(8)))  float  v8f;

union FragBF { uint4 q[2]; v16bf v; };

// packed f32x2 -> bf16x2; prefer dedicated builtin, else native __bf16 cast
// (legal fptrunc on gfx1250 -> hardware cvt), never manual bit-twiddling.
__device__ __forceinline__ u32 pk2bf(float a, float b) {
#if __has_builtin(__builtin_amdgcn_cvt_pk_bf16_f32)
    auto r = __builtin_amdgcn_cvt_pk_bf16_f32(a, b);
    u32 u;
    __builtin_memcpy(&u, &r, 4);
    return u;
#else
    __bf16 lo = (__bf16)a;
    __bf16 hi = (__bf16)b;
    unsigned short ulo, uhi;
    __builtin_memcpy(&ulo, &lo, 2);
    __builtin_memcpy(&uhi, &hi, 2);
    return (u32)ulo | ((u32)uhi << 16);
#endif
}

// ---------------------------------------------------------------------------
// Pre-pack a row-major f32 weight W[K][N] into bf16 WMMA B-fragment order:
// dword id = ((nt*KT + kt)*32 + lane)*8 + v  holds  {W[k][n], W[k+1][n]}
// with k = kt*32 + (lane>=16 ? 16:0) + 2v, n = nt*16 + (lane&15).
// Each lane then loads a full v16bf fragment as two contiguous uint4.
// ---------------------------------------------------------------------------
__global__ void prepack_kernel(const float* __restrict__ W, u32* __restrict__ dst,
                               int K, int N) {
    int id = blockIdx.x * 256 + threadIdx.x;
    int KT = K >> 5;
    int total = (N >> 4) * KT * 256;
    if (id >= total) return;
    int v    = id & 7;
    int lane = (id >> 3) & 31;
    int rest = id >> 8;
    int kt = rest % KT;
    int nt = rest / KT;
    int k = kt * 32 + ((lane & 16) ? 16 : 0) + 2 * v;
    int n = nt * 16 + (lane & 15);
    dst[id] = pk2bf(W[k * N + n], W[(k + 1) * N + n]);
}

// Build the A fragment (16x32 bf16 tile, M rows in LDS with row stride `stride`)
__device__ __forceinline__ FragBF loadA(const unsigned short* s, int lane,
                                        int kt, int stride) {
    int m    = lane & 15;
    int koff = (lane >> 4) << 4;            // 0 or 16
    const uint4* p = (const uint4*)(s + m * stride + kt * 32 + koff);
    FragBF a;
    a.q[0] = p[0];                          // K = base .. base+7
    a.q[1] = p[1];                          // K = base+8 .. base+15
    return a;
}

__device__ __forceinline__ FragBF loadB(const u32* __restrict__ Wp,
                                        int nt, int KT, int kt, int lane) {
    const uint4* p = (const uint4*)(Wp + (((nt * KT) + kt) * 32 + lane) * 8);
    FragBF b;
    b.q[0] = p[0];
    b.q[1] = p[1];
    return b;
}

// One [16 rows]x[16 cols] layer tile (K=128) with relu, LDS->LDS (bf16)
__device__ __forceinline__ void layer128(const unsigned short* sIn,
                                         unsigned short* sOut,
                                         const u32* __restrict__ Wp,
                                         const float* __restrict__ bias,
                                         int lane, int w) {
    int ncol = w * 16 + (lane & 15);
    float bv = bias[ncol];
    v8f acc;
#pragma unroll
    for (int j = 0; j < 8; ++j) acc[j] = bv;
#pragma unroll
    for (int kt = 0; kt < 4; ++kt) {
        FragBF a = loadA(sIn, lane, kt, 128);
        FragBF b = loadB(Wp, w, 4, kt, lane);
        acc = __builtin_amdgcn_wmma_f32_16x16x32_bf16(false, a.v, false, b.v,
                                                      (short)0, acc, false, false);
    }
    int mbase = (lane >> 4) << 3;
#pragma unroll
    for (int j = 0; j < 8; j += 2) {
        u32 p = pk2bf(fmaxf(acc[j], 0.f), fmaxf(acc[j + 1], 0.f));
        sOut[(j + mbase) * 128 + ncol]     = (unsigned short)(p & 0xFFFFu);
        sOut[(j + 1 + mbase) * 128 + ncol] = (unsigned short)(p >> 16);
    }
}

// ---------------------------------------------------------------------------
// Kernel A: a = h + c*kprev; z3 = relu(relu(relu(a@Win+b)@Wh1+b)@Wh2+b)
// grid 32 (16 batch rows each), 256 threads (8 waves = 8 column tiles)
// ---------------------------------------------------------------------------
__global__ void __launch_bounds__(256)
mlp3_kernel(const float* h, const float* kprev, float c,
            const u32* __restrict__ WpIn, const float* __restrict__ bIn,
            const u32* __restrict__ Wp1,  const float* __restrict__ b1,
            const u32* __restrict__ Wp2,  const float* __restrict__ b2,
            unsigned short* __restrict__ z3) {
    __shared__ unsigned short sA[16 * 128];
    __shared__ unsigned short sB[16 * 128];
    int tid = threadIdx.x, lane = tid & 31, w = tid >> 5;
    int r0 = blockIdx.x * 16;

    // 8 contiguous elements per thread: float4 loads + packed bf16 converts
    {
        int base = r0 * 128 + tid * 8;
        float4 h0 = *(const float4*)(h + base);
        float4 h1 = *(const float4*)(h + base + 4);
        float4 p0 = *(const float4*)(kprev + base);
        float4 p1 = *(const float4*)(kprev + base + 4);
        u32* dst = (u32*)sA + tid * 4;
        dst[0] = pk2bf(h0.x + c * p0.x, h0.y + c * p0.y);
        dst[1] = pk2bf(h0.z + c * p0.z, h0.w + c * p0.w);
        dst[2] = pk2bf(h1.x + c * p1.x, h1.y + c * p1.y);
        dst[3] = pk2bf(h1.z + c * p1.z, h1.w + c * p1.w);
    }
    __syncthreads();
    layer128(sA, sB, WpIn, bIn, lane, w);
    __syncthreads();
    layer128(sB, sA, Wp1, b1, lane, w);
    __syncthreads();
    layer128(sA, sB, Wp2, b2, lane, w);
    __syncthreads();
    ((uint4*)(z3 + r0 * 128))[tid] = ((const uint4*)sB)[tid];
}

// ---------------------------------------------------------------------------
// Kernel B: F = tanh(z3 @ Wout + bout) [512,4096]; k[b,h] = sum_i F[b,h,i]*d[b,i]
// grid (8 col blocks of 512, 32 row blocks of 16), 256 threads.
// Each wave owns 4 N-tiles = 2 h values (full i range) -> shfl reduction,
// exactly one global write per (b,h).
// ---------------------------------------------------------------------------
__global__ void __launch_bounds__(256)
bigvf_kernel(const float* __restrict__ x, int t,
             const unsigned short* __restrict__ z3,
             const u32* __restrict__ WpOut,
             const float* __restrict__ bOut,
             float* __restrict__ kout) {
    __shared__ unsigned short sA[16 * 128];
    __shared__ float sD[16 * 32];
    int tid = threadIdx.x, lane = tid & 31, w = tid >> 5;
    int cb = blockIdx.x;
    int r0 = blockIdx.y * 16;

    ((uint4*)sA)[tid] = ((const uint4*)(z3 + r0 * 128))[tid];
    for (int e = tid; e < 512; e += 256) {
        int m = e >> 5, i = e & 31;
        int b = r0 + m;
        sD[e] = x[(b * T_ + (t + 1)) * IN_ + i] - x[(b * T_ + t) * IN_ + i];
    }
    __syncthreads();

    FragBF af[4];
#pragma unroll
    for (int kt = 0; kt < 4; ++kt) af[kt] = loadA(sA, lane, kt, 128);

    int mbase = (lane >> 4) << 3;
    float kpart[8];
#pragma unroll
    for (int j4 = 0; j4 < 4; ++j4) {
        int tl = w * 4 + j4;          // local tile 0..31 (parity == j4 parity)
        int nt = cb * 32 + tl;        // global N tile 0..255
        if (j4 < 3)
            __builtin_prefetch(WpOut + (((nt + 1) * 4) * 32 + lane) * 8, 0, 1);
        float bv = bOut[nt * 16 + (lane & 15)];
        v8f acc;
#pragma unroll
        for (int j = 0; j < 8; ++j) acc[j] = bv;
#pragma unroll
        for (int kt = 0; kt < 4; ++kt) {
            FragBF b = loadB(WpOut, nt, 4, kt, lane);
            acc = __builtin_amdgcn_wmma_f32_16x16x32_bf16(false, af[kt].v, false, b.v,
                                                          (short)0, acc, false, false);
        }
        int hl = tl >> 1;
        int i  = ((tl & 1) << 4) + (lane & 15);
#pragma unroll
        for (int j = 0; j < 8; ++j) {
            float f = tanhf(acc[j]);
            float p = f * sD[(j + mbase) * 32 + i];
            p += __shfl_xor(p, 1, 32);
            p += __shfl_xor(p, 2, 32);
            p += __shfl_xor(p, 4, 32);
            p += __shfl_xor(p, 8, 32);   // lanes 0 / 16 hold the i-halves' sums
            if ((j4 & 1) == 0) {
                kpart[j] = p;            // even tile: i = 0..15 partial
            } else {
                if ((lane & 15) == 0)
                    kout[(r0 + j + mbase) * 128 + blockIdx.x * 16 + hl] = kpart[j] + p;
            }
        }
    }
}

// ---------------------------------------------------------------------------
// Kernel C: h = h + (k1 + 2(k2+k3) + k4)/6  (skipped when k1==nullptr),
// then out[:, t_out, :] = h @ Wfin + bfin   (WMMA, 4 col tiles on waves 0..3)
// ---------------------------------------------------------------------------
__global__ void __launch_bounds__(256)
combine_out_kernel(float* h,
                   const float* k1, const float* k2,
                   const float* k3, const float* k4,
                   const u32* __restrict__ WpFin,
                   const float* __restrict__ bFin,
                   float* __restrict__ out, int t_out) {
    __shared__ unsigned short sA[16 * 128];
    int tid = threadIdx.x, lane = tid & 31, w = tid >> 5;
    int r0 = blockIdx.x * 16;

    {
        int base = r0 * 128 + tid * 8;
        float4 h0 = *(const float4*)(h + base);
        float4 h1 = *(const float4*)(h + base + 4);
        if (k1) {
            const float s = 1.f / 6.f;
            float4 a0 = *(const float4*)(k1 + base), a1 = *(const float4*)(k1 + base + 4);
            float4 b0 = *(const float4*)(k2 + base), b1 = *(const float4*)(k2 + base + 4);
            float4 c0 = *(const float4*)(k3 + base), c1 = *(const float4*)(k3 + base + 4);
            float4 d0 = *(const float4*)(k4 + base), d1 = *(const float4*)(k4 + base + 4);
            h0.x += (a0.x + 2.f * (b0.x + c0.x) + d0.x) * s;
            h0.y += (a0.y + 2.f * (b0.y + c0.y) + d0.y) * s;
            h0.z += (a0.z + 2.f * (b0.z + c0.z) + d0.z) * s;
            h0.w += (a0.w + 2.f * (b0.w + c0.w) + d0.w) * s;
            h1.x += (a1.x + 2.f * (b1.x + c1.x) + d1.x) * s;
            h1.y += (a1.y + 2.f * (b1.y + c1.y) + d1.y) * s;
            h1.z += (a1.z + 2.f * (b1.z + c1.z) + d1.z) * s;
            h1.w += (a1.w + 2.f * (b1.w + c1.w) + d1.w) * s;
            *(float4*)(h + base)     = h0;
            *(float4*)(h + base + 4) = h1;
        }
        u32* dst = (u32*)sA + tid * 4;
        dst[0] = pk2bf(h0.x, h0.y);
        dst[1] = pk2bf(h0.z, h0.w);
        dst[2] = pk2bf(h1.x, h1.y);
        dst[3] = pk2bf(h1.z, h1.w);
    }
    __syncthreads();

    if (w < 4) {
        float bv = bFin[w * 16 + (lane & 15)];
        v8f acc;
#pragma unroll
        for (int j = 0; j < 8; ++j) acc[j] = bv;
#pragma unroll
        for (int kt = 0; kt < 4; ++kt) {
            FragBF a = loadA(sA, lane, kt, 128);
            FragBF b = loadB(WpFin, w, 4, kt, lane);
            acc = __builtin_amdgcn_wmma_f32_16x16x32_bf16(false, a.v, false, b.v,
                                                          (short)0, acc, false, false);
        }
        int mbase = (lane >> 4) << 3;
#pragma unroll
        for (int j = 0; j < 8; ++j)
            out[((long)(r0 + j + mbase) * T_ + t_out) * OUT_ + w * 16 + (lane & 15)] = acc[j];
    }
}

// ---------------------------------------------------------------------------
// h0 = x[:,0,:] @ W_init + b_init   ([512,32]@[32,128], single K-step WMMA)
// ---------------------------------------------------------------------------
__global__ void __launch_bounds__(256)
init_h0_kernel(const float* __restrict__ x,
               const u32* __restrict__ WpInit,
               const float* __restrict__ bInit,
               float* __restrict__ h) {
    __shared__ unsigned short sA[16 * 32];
    int tid = threadIdx.x, lane = tid & 31, w = tid >> 5;
    int r0 = blockIdx.x * 16;
    {
        int m = tid >> 4, i2 = (tid & 15) * 2;
        const float* px = x + ((long)(r0 + m) * T_) * IN_ + i2;
        ((u32*)sA)[tid] = pk2bf(px[0], px[1]);
    }
    __syncthreads();
    FragBF a = loadA(sA, lane, 0, 32);
    FragBF b = loadB(WpInit, w, 1, 0, lane);
    float bv = bInit[w * 16 + (lane & 15)];
    v8f acc;
#pragma unroll
    for (int j = 0; j < 8; ++j) acc[j] = bv;
    acc = __builtin_amdgcn_wmma_f32_16x16x32_bf16(false, a.v, false, b.v,
                                                  (short)0, acc, false, false);
    int mbase = (lane >> 4) << 3;
#pragma unroll
    for (int j = 0; j < 8; ++j)
        h[(r0 + j + mbase) * 128 + w * 16 + (lane & 15)] = acc[j];
}

// ---------------------------------------------------------------------------
extern "C" void kernel_launch(void* const* d_in, const int* in_sizes, int n_in,
                              void* d_out, int out_size, void* d_ws, size_t ws_size,
                              hipStream_t stream) {
    const float* x      = (const float*)d_in[0];
    const float* W_init = (const float*)d_in[1];
    const float* b_init = (const float*)d_in[2];
    const float* W_in   = (const float*)d_in[3];
    const float* b_in   = (const float*)d_in[4];
    const float* W_h1   = (const float*)d_in[5];
    const float* b_h1   = (const float*)d_in[6];
    const float* W_h2   = (const float*)d_in[7];
    const float* b_h2   = (const float*)d_in[8];
    const float* W_out  = (const float*)d_in[9];
    const float* b_out  = (const float*)d_in[10];
    const float* W_fin  = (const float*)d_in[11];
    const float* b_fin  = (const float*)d_in[12];
    float* out = (float*)d_out;

    // workspace layout
    u32* Wp_init = (u32*)d_ws;                 //   2048 dw
    u32* Wp_in   = Wp_init + 2048;             //   8192 dw
    u32* Wp_h1   = Wp_in   + 8192;
    u32* Wp_h2   = Wp_h1   + 8192;
    u32* Wp_out  = Wp_h2   + 8192;             // 262144 dw
    u32* Wp_fin  = Wp_out  + 262144;           //   4096 dw
    float* h  = (float*)(Wp_fin + 4096);       // 65536 f32
    float* k1 = h  + 65536;
    float* k2 = k1 + 65536;
    float* k3 = k2 + 65536;
    float* k4 = k3 + 65536;
    unsigned short* z3 = (unsigned short*)(k4 + 65536);  // 65536 bf16

    // one-time weight packing (bf16 fragment order)
    prepack_kernel<<<8,    256, 0, stream>>>(W_init, Wp_init, 32, 128);
    prepack_kernel<<<32,   256, 0, stream>>>(W_in,  Wp_in,  128, 128);
    prepack_kernel<<<32,   256, 0, stream>>>(W_h1,  Wp_h1,  128, 128);
    prepack_kernel<<<32,   256, 0, stream>>>(W_h2,  Wp_h2,  128, 128);
    prepack_kernel<<<1024, 256, 0, stream>>>(W_out, Wp_out, 128, 4096);
    prepack_kernel<<<16,   256, 0, stream>>>(W_fin, Wp_fin, 128, 64);

    init_h0_kernel<<<32, 256, 0, stream>>>(x, Wp_init, b_init, h);
    combine_out_kernel<<<32, 256, 0, stream>>>(h, nullptr, nullptr, nullptr, nullptr,
                                               Wp_fin, b_fin, out, 0);

    dim3 gB(8, 32);
    for (int t = 0; t < T_ - 1; ++t) {
        mlp3_kernel<<<32, 256, 0, stream>>>(h, h, 0.0f, Wp_in, b_in, Wp_h1, b_h1,
                                            Wp_h2, b_h2, z3);
        bigvf_kernel<<<gB, 256, 0, stream>>>(x, t, z3, Wp_out, b_out, k1);

        mlp3_kernel<<<32, 256, 0, stream>>>(h, k1, 0.5f, Wp_in, b_in, Wp_h1, b_h1,
                                            Wp_h2, b_h2, z3);
        bigvf_kernel<<<gB, 256, 0, stream>>>(x, t, z3, Wp_out, b_out, k2);

        mlp3_kernel<<<32, 256, 0, stream>>>(h, k2, 0.5f, Wp_in, b_in, Wp_h1, b_h1,
                                            Wp_h2, b_h2, z3);
        bigvf_kernel<<<gB, 256, 0, stream>>>(x, t, z3, Wp_out, b_out, k3);

        mlp3_kernel<<<32, 256, 0, stream>>>(h, k3, 1.0f, Wp_in, b_in, Wp_h1, b_h1,
                                            Wp_h2, b_h2, z3);
        bigvf_kernel<<<gB, 256, 0, stream>>>(x, t, z3, Wp_out, b_out, k4);

        combine_out_kernel<<<32, 256, 0, stream>>>(h, k1, k2, k3, k4,
                                                   Wp_fin, b_fin, out, t + 1);
    }
}
